// PolyHashV12_42606075576732
// MI455X (gfx1250) — matrix operand order repaired and checked by reading, not compiled, not run on hardware
//
#include <hip/hip_runtime.h>
#include <stdint.h>

// ---------------- model constants ----------------
#define NUM_TABLES 8
#define BUCKETSN   65536
#define EMBEDN     64
#define DIMD       512        // NUM_TABLES * EMBED
#define SCAND      1024
#define HEADSN     8
#define HDN        64
#define BSZ        2
#define TLEN       2048
#define LN_EPS     1e-5f

// ---------------- WMMA types ----------------
typedef __attribute__((ext_vector_type(16))) unsigned short u16x16;
typedef __attribute__((ext_vector_type(16))) __bf16         bf16x16;
typedef __attribute__((ext_vector_type(8)))  float          f32x8;

union Frag16 { u16x16 u; bf16x16 b; };

static __device__ __forceinline__ unsigned short f2bf(float f) {
  unsigned int u = __float_as_uint(f);
  u += 0x7fffu + ((u >> 16) & 1u);          // round-to-nearest-even
  return (unsigned short)(u >> 16);
}

static __device__ __forceinline__ f32x8 wmma_bf16(const Frag16& a, const Frag16& b, f32x8 c) {
  // D(16x16,f32) = A(16x32,bf16) * B(32x16,bf16) + C
  return __builtin_amdgcn_wmma_f32_16x16x32_bf16(false, a.b, false, b.b, (short)0, c, false, false);
}

// ---------------- elementwise: f32 -> bf16 ----------------
__global__ void f32_to_bf16_kernel(const float* __restrict__ src,
                                   unsigned short* __restrict__ dst, int n) {
  int i = blockIdx.x * blockDim.x + threadIdx.x;
  if (i < n) dst[i] = f2bf(src[i]);
}

// ---------------- hash indices ----------------
__global__ void hash_idx_kernel(const long long* __restrict__ tokens,
                                int* __restrict__ idx, int Bn, int T) {
  int i = blockIdx.x * blockDim.x + threadIdx.x;
  if (i >= Bn * T) return;
  int b = i / T, t = i % T;
  const long long* tb = tokens + (size_t)b * T;
  const unsigned long long primes[8] = {2654435761ull, 2246822519ull, 3266489917ull,
                                        2028178513ull, 1220703125ull, 1610612741ull,
                                        805306457ull,  402653189ull};
  unsigned long long acc = 0;
  int tc = 0;
  for (int j = 0; j < 128; ++j) {
    int offn = j + 1;
    if (offn >= T) break;
    if (t - offn >= 0)
      acc ^= (unsigned long long)tb[t - offn] * primes[j & 7];
    if ((offn & (offn - 1)) == 0) {          // window in {1,2,4,...,128}
      idx[(size_t)i * NUM_TABLES + tc] = (int)(acc % BUCKETSN);
      ++tc;
    }
  }
}

// ---------------- gather embeddings ----------------
__global__ void gather_embed_kernel(const float* __restrict__ tables,
                                    const int* __restrict__ idx,
                                    float* __restrict__ x0,
                                    unsigned short* __restrict__ x0bf, int total) {
  int i = blockIdx.x * blockDim.x + threadIdx.x;
  if (i >= total) return;
  int e   = i & 63;
  int tbl = (i >> 6) & 7;
  int bt  = i >> 9;
  int id  = idx[(size_t)bt * NUM_TABLES + tbl];
  float v = tables[((size_t)tbl * BUCKETSN + id) * EMBEDN + e];
  x0[i]   = v;
  x0bf[i] = f2bf(v);
}

// ---------------- GEMM: C[M,N] f32 = A[M,K]bf16 @ W[K,N]bf16 + bias ----------------
// block = 256 threads = 8 waves; C tile 128x64; each wave a 16x64 strip:
// one A fragment reused across 4 back-to-back WMMAs per K-step (K step 32).
__global__ void gemm_bf16_wmma_kernel(const unsigned short* __restrict__ A,
                                      const unsigned short* __restrict__ W,
                                      const float* __restrict__ bias,
                                      float* __restrict__ C,
                                      int M, int N, int K) {
  __shared__ unsigned short As[128][32];     // 8 KB
  __shared__ unsigned short Bs[64][34];      // transposed, padded
  const int tid  = threadIdx.x;
  const int wave = tid >> 5;
  const int lane = tid & 31;
  const int m0 = blockIdx.y * 128;
  const int n0 = blockIdx.x * 64;
  const int waveM = wave * 16;

  f32x8 acc[4];
  #pragma unroll
  for (int g = 0; g < 4; ++g)
    #pragma unroll
    for (int j = 0; j < 8; ++j) acc[g][j] = 0.f;

  const int ar  = tid >> 1;                  // A tile row 0..127
  const int ac  = (tid & 1) * 16;            // A tile k   0 / 16
  const int bk  = tid >> 3;                  // B tile k row 0..31
  const int bn  = (tid & 7) * 8;             // B tile n offset 0..56

  for (int k0 = 0; k0 < K; k0 += 32) {
    // stage A tile (128x32 halves, 16B coalesced per thread)
    const unsigned short* asrc = A + (size_t)(m0 + ar) * K + k0 + ac;
    *(uint4*)(&As[ar][ac]) = *(const uint4*)asrc;
    // stage B tile: one coalesced 16B row chunk, scatter-transpose into LDS
    const unsigned short* bsrc = W + (size_t)(k0 + bk) * N + n0 + bn;
    {
      unsigned short tmp[8];
      *(uint4*)tmp = *(const uint4*)bsrc;
      #pragma unroll
      for (int i = 0; i < 8; ++i)
        Bs[bn + i][bk] = tmp[i];
    }
    if (k0 + 32 < K) {
      __builtin_prefetch(asrc + 32, 0, 0);
      __builtin_prefetch(bsrc + (size_t)32 * N, 0, 0);
    }
    __syncthreads();

    // one A fragment per wave, reused for 4 WMMAs
    Frag16 af;
    const int row   = waveM + (lane & 15);
    const int koffA = (lane >> 4) * 8;
    #pragma unroll
    for (int e = 0; e < 16; ++e)
      af.u[e] = As[row][koffA + ((e & 8) ? 16 : 0) + (e & 7)];

    const int koffB = (lane >> 4) * 8;
    #pragma unroll
    for (int g = 0; g < 4; ++g) {
      Frag16 bf;
      const int col = g * 16 + (lane & 15);
      #pragma unroll
      for (int e = 0; e < 16; ++e)
        bf.u[e] = Bs[col][koffB + ((e & 8) ? 16 : 0) + (e & 7)];
      acc[g] = wmma_bf16(af, bf, acc[g]);
    }
    __syncthreads();
  }

  #pragma unroll
  for (int g = 0; g < 4; ++g) {
    const int col = n0 + g * 16 + (lane & 15);
    const float bv = bias ? bias[col] : 0.f;
    #pragma unroll
    for (int j = 0; j < 8; ++j) {
      const int rowg = m0 + waveM + j + ((lane >> 4) * 8);
      C[(size_t)rowg * N + col] = acc[g][j] + bv;
    }
  }
}

// ---------------- scan elementwise: gates / drive ----------------
__global__ void scan_gates_kernel(const float* __restrict__ zg,
                                  const float* __restrict__ zdt,
                                  const float* __restrict__ zin,
                                  float* __restrict__ gates,
                                  float* __restrict__ drive, int n) {
  int i = blockIdx.x * blockDim.x + threadIdx.x;
  if (i >= n) return;
  float d  = zdt[i];
  float sp = (d > 20.f) ? d : log1pf(__expf(d));         // softplus
  float g  = 1.f / (1.f + __expf(-zg[i] * sp));          // sigmoid(z*dt)
  gates[i] = g;
  drive[i] = (1.f - g) * zin[i];
}

// ---------------- sequential scan h = clamp(a)*h + d ----------------
__global__ void scan_seq_kernel(const float* __restrict__ gates,
                                const float* __restrict__ drive,
                                unsigned short* __restrict__ states_bf,
                                int Bn, int T, int D) {
  int idx = blockIdx.x * blockDim.x + threadIdx.x;
  if (idx >= Bn * D) return;
  int b = idx / D, d = idx % D;
  const float* g  = gates + (size_t)b * T * D + d;
  const float* dr = drive + (size_t)b * T * D + d;
  unsigned short* o = states_bf + (size_t)b * T * D + d;
  float h = 0.f;
  for (int t = 0; t < T; ++t) {
    float a = fmaxf(g[(size_t)t * D], 1e-6f);
    h = a * h + dr[(size_t)t * D];
    o[(size_t)t * D] = f2bf(h);
  }
}

// ---------------- fused residual + LayerNorm (D = 512) ----------------
__global__ void ln_fused_kernel(const float* __restrict__ res,
                                const float* __restrict__ delta,
                                const float* __restrict__ gamma,
                                const float* __restrict__ beta,
                                float* __restrict__ out,
                                unsigned short* __restrict__ outbf) {
  __shared__ float red[256];
  const int D = DIMD;
  const int row = blockIdx.x, tid = threadIdx.x;
  const size_t base = (size_t)row * D;
  float v0 = res[base + tid]       + delta[base + tid];
  float v1 = res[base + tid + 256] + delta[base + tid + 256];
  red[tid] = v0 + v1;
  __syncthreads();
  for (int s = 128; s > 0; s >>= 1) { if (tid < s) red[tid] += red[tid + s]; __syncthreads(); }
  float mean = red[0] * (1.0f / D);
  __syncthreads();
  float d0 = v0 - mean, d1 = v1 - mean;
  red[tid] = d0 * d0 + d1 * d1;
  __syncthreads();
  for (int s = 128; s > 0; s >>= 1) { if (tid < s) red[tid] += red[tid + s]; __syncthreads(); }
  float inv = rsqrtf(red[0] * (1.0f / D) + LN_EPS);
  float y0 = d0 * inv * gamma[tid]       + beta[tid];
  float y1 = d1 * inv * gamma[tid + 256] + beta[tid + 256];
  out[base + tid]       = y0;
  out[base + tid + 256] = y1;
  if (outbf) { outbf[base + tid] = f2bf(y0); outbf[base + tid + 256] = f2bf(y1); }
}

// ---------------- rearrange (rows, srow) f32 -> (B,H,T,HD) bf16 ----------------
__global__ void split_bhtd_kernel(const float* __restrict__ src,
                                  unsigned short* __restrict__ dst,
                                  int T, int srow, int coff) {
  int i = blockIdx.x * blockDim.x + threadIdx.x;
  int total = BSZ * HEADSN * T * HDN;
  if (i >= total) return;
  int d = i & 63;
  int t = (i >> 6) % T;
  int h = (i / (HDN * T)) % HEADSN;
  int b = i / (HDN * T * HEADSN);
  dst[i] = f2bf(src[((size_t)(b * T + t)) * srow + coff + h * HDN + d]);
}

// ---------------- replicate inducing points -> (B*M_PTS, DIM) bf16 ----------------
__global__ void replicate_ind_kernel(const float* __restrict__ ind,
                                     unsigned short* __restrict__ dst, int total) {
  int i = blockIdx.x * blockDim.x + threadIdx.x;
  if (i >= total) return;
  int c = i & (DIMD - 1);
  int r = i >> 9;
  dst[i] = f2bf(ind[(size_t)(r & 63) * DIMD + c]);
}

// ---------------- flash attention, 1 wave per (b,h,16-row q tile) ----------------
// Q/K/V bf16 laid out (B,H,T,64); output bf16 (B,Tq,H*64).
__global__ void flash_attn_kernel(const unsigned short* __restrict__ Qb,
                                  const unsigned short* __restrict__ Kb,
                                  const unsigned short* __restrict__ Vb,
                                  unsigned short* __restrict__ Ob,
                                  int Tq, int Tk, int causal, float scale) {
  __shared__ float          s_s[16][33];
  __shared__ unsigned short p_s[16][36];
  __shared__ unsigned short v_s[32][64];     // staged V tile (coalesced loads)
  __shared__ float m_s[16], l_s[16], c_s[16];

  const int lane = threadIdx.x;
  const int q0 = blockIdx.x * 16;
  const int h  = blockIdx.y;
  const int b  = blockIdx.z;
  const unsigned short* qbase = Qb + ((size_t)(b * HEADSN + h) * Tq) * HDN;
  const unsigned short* kbase = Kb + ((size_t)(b * HEADSN + h) * Tk) * HDN;
  const unsigned short* vbase = Vb + ((size_t)(b * HEADSN + h) * Tk) * HDN;
  const float NEG_INF = -__builtin_inff();

  // Q fragments: two 16x32 A-tiles covering d = 0..63
  Frag16 qf[2];
  {
    const int row  = q0 + (lane & 15);
    const int koff = (lane >> 4) * 8;
    #pragma unroll
    for (int d32 = 0; d32 < 2; ++d32)
      #pragma unroll
      for (int e = 0; e < 16; ++e) {
        int d = d32 * 32 + koff + ((e & 8) ? 16 : 0) + (e & 7);
        qf[d32].u[e] = qbase[(size_t)row * HDN + d];
      }
  }
  if (lane < 16) { m_s[lane] = NEG_INF; l_s[lane] = 0.f; }

  f32x8 o[4];
  #pragma unroll
  for (int g = 0; g < 4; ++g)
    #pragma unroll
    for (int j = 0; j < 8; ++j) o[g][j] = 0.f;

  const int nkt = causal ? ((q0 + 16 + 31) / 32) : (Tk / 32);
  for (int kt = 0; kt < nkt; ++kt) {
    const int kb = kt * 32;
    // S tile (16 x 32) = Q (16x64) @ K^T, two 16-col WMMA subtiles
    #pragma unroll
    for (int sub = 0; sub < 2; ++sub) {
      f32x8 s;
      #pragma unroll
      for (int j = 0; j < 8; ++j) s[j] = 0.f;
      #pragma unroll
      for (int d32 = 0; d32 < 2; ++d32) {
        Frag16 kf;
        const int kcol = kb + sub * 16 + (lane & 15);
        const int dof  = d32 * 32 + (lane >> 4) * 8;
        #pragma unroll
        for (int e = 0; e < 16; ++e)
          kf.u[e] = kbase[(size_t)kcol * HDN + dof + ((e & 8) ? 16 : 0) + (e & 7)];
        s = wmma_bf16(qf[d32], kf, s);
      }
      #pragma unroll
      for (int j = 0; j < 8; ++j)
        s_s[j + ((lane >> 4) * 8)][sub * 16 + (lane & 15)] = s[j];
    }

    // stage V tile (32 k-rows x 64 d) -- one coalesced row per lane
    {
      const unsigned short* vsrc = vbase + (size_t)(kb + lane) * HDN;
      #pragma unroll
      for (int q = 0; q < 4; ++q)
        *(uint4*)(&v_s[lane][q * 16]) = *(const uint4*)(vsrc + q * 16);
    }
    __syncthreads();

    // online softmax row update (rows 0..15 on lanes 0..15)
    if (lane < 16) {
      const int r = lane;
      float mOld = m_s[r];
      float mNew = mOld;
      for (int c = 0; c < 32; ++c) {
        int kidx = kb + c;
        bool ok = (kidx < Tk) && (!causal || kidx <= q0 + r);
        float v = ok ? s_s[r][c] * scale : NEG_INF;
        mNew = fmaxf(mNew, v);
      }
      float corr = (mOld == NEG_INF) ? 0.f : __expf(mOld - mNew);
      float rs = 0.f;
      for (int c = 0; c < 32; ++c) {
        int kidx = kb + c;
        bool ok = (kidx < Tk) && (!causal || kidx <= q0 + r);
        float p = ok ? __expf(s_s[r][c] * scale - mNew) : 0.f;
        rs += p;
        p_s[r][c] = f2bf(p);
      }
      m_s[r] = mNew;
      l_s[r] = l_s[r] * corr + rs;
      c_s[r] = corr;
    }
    __syncthreads();

    // rescale running output rows
    #pragma unroll
    for (int j = 0; j < 8; ++j) {
      float cr = c_s[j + ((lane >> 4) * 8)];
      #pragma unroll
      for (int g = 0; g < 4; ++g) o[g][j] *= cr;
    }

    // O += P (16x32) @ V (32x64); V fragments from LDS
    Frag16 pf;
    {
      const int row  = lane & 15;
      const int koff = (lane >> 4) * 8;
      #pragma unroll
      for (int e = 0; e < 16; ++e)
        pf.u[e] = p_s[row][koff + ((e & 8) ? 16 : 0) + (e & 7)];
    }
    const int klo = (lane >> 4) * 8;
    #pragma unroll
    for (int g = 0; g < 4; ++g) {
      Frag16 vf;
      const int dcol = g * 16 + (lane & 15);
      #pragma unroll
      for (int e = 0; e < 16; ++e)
        vf.u[e] = v_s[klo + ((e & 8) ? 16 : 0) + (e & 7)][dcol];
      o[g] = wmma_bf16(pf, vf, o[g]);
    }
    __syncthreads();
  }

  if (lane < 16) c_s[lane] = (l_s[lane] > 0.f) ? (1.f / l_s[lane]) : 0.f;
  __syncthreads();
  const int dlane = lane & 15;
  #pragma unroll
  for (int j = 0; j < 8; ++j) {
    const int r = j + ((lane >> 4) * 8);
    const float inv = c_s[r];
    const int t = q0 + r;
    #pragma unroll
    for (int g = 0; g < 4; ++g)
      Ob[((size_t)(b * Tq + t)) * DIMD + h * HDN + g * 16 + dlane] = f2bf(o[g][j] * inv);
  }
}

// ---------------- host orchestration ----------------
extern "C" void kernel_launch(void* const* d_in, const int* in_sizes, int n_in,
                              void* d_out, int out_size, void* d_ws, size_t ws_size,
                              hipStream_t stream) {
  (void)in_sizes; (void)n_in; (void)out_size; (void)ws_size;
  const float* tables     = (const float*)d_in[0];
  const float* gate_W     = (const float*)d_in[1];
  const float* gate_b     = (const float*)d_in[2];
  const float* dt_W       = (const float*)d_in[3];
  const float* dt_b       = (const float*)d_in[4];
  const float* in_W       = (const float*)d_in[5];
  const float* in_b       = (const float*)d_in[6];
  const float* scan_out_W = (const float*)d_in[7];
  const float* scan_out_b = (const float*)d_in[8];
  const float* scan_ln_g  = (const float*)d_in[9];
  const float* scan_ln_b  = (const float*)d_in[10];
  const float* qkv_W      = (const float*)d_in[11];
  const float* qkv_b      = (const float*)d_in[12];
  const float* attn_out_W = (const float*)d_in[13];
  const float* attn_out_b = (const float*)d_in[14];
  const float* attn_ln_g  = (const float*)d_in[15];
  const float* attn_ln_b  = (const float*)d_in[16];
  const float* inducing   = (const float*)d_in[17];
  const float* qd_W = (const float*)d_in[18];
  const float* qd_b = (const float*)d_in[19];
  const float* kd_W = (const float*)d_in[20];
  const float* kd_b = (const float*)d_in[21];
  const float* vd_W = (const float*)d_in[22];
  const float* vd_b = (const float*)d_in[23];
  const float* qu_W = (const float*)d_in[24];
  const float* qu_b = (const float*)d_in[25];
  const float* ku_W = (const float*)d_in[26];
  const float* ku_b = (const float*)d_in[27];
  const float* vu_W = (const float*)d_in[28];
  const float* vu_b = (const float*)d_in[29];
  const float* isab_out_W = (const float*)d_in[30];
  const float* isab_out_b = (const float*)d_in[31];
  const float* isab_ln_g  = (const float*)d_in[32];
  const float* isab_ln_b  = (const float*)d_in[33];
  const long long* tokens = (const long long*)d_in[34];   // jnp.int64
  float* out = (float*)d_out;

  const int BT = BSZ * TLEN;               // 4096 rows
  const int NBHTD = BSZ * HEADSN * TLEN * HDN;
  const int NBHMD = BSZ * HEADSN * 64 * HDN;

  size_t off = 0;
  auto alloc  = [&](size_t bytes) -> void* {
    off = (off + 255) & ~(size_t)255;
    void* p = (char*)d_ws + off;
    off += bytes;
    return p;
  };
  auto allocF = [&](size_t n) { return (float*)alloc(n * sizeof(float)); };
  auto allocH = [&](size_t n) { return (unsigned short*)alloc(n * sizeof(unsigned short)); };
  auto allocI = [&](size_t n) { return (int*)alloc(n * sizeof(int)); };

  // bf16 weights
  unsigned short* wbGate = allocH((size_t)DIMD * SCAND);
  unsigned short* wbDt   = allocH((size_t)DIMD * SCAND);
  unsigned short* wbIn   = allocH((size_t)DIMD * SCAND);
  unsigned short* wbScan = allocH((size_t)SCAND * DIMD);
  unsigned short* wbQkv  = allocH((size_t)DIMD * 3 * DIMD);
  unsigned short* wbAttn = allocH((size_t)DIMD * DIMD);
  unsigned short* wbQd   = allocH((size_t)DIMD * DIMD);
  unsigned short* wbKd   = allocH((size_t)DIMD * DIMD);
  unsigned short* wbVd   = allocH((size_t)DIMD * DIMD);
  unsigned short* wbQu   = allocH((size_t)DIMD * DIMD);
  unsigned short* wbKu   = allocH((size_t)DIMD * DIMD);
  unsigned short* wbVu   = allocH((size_t)DIMD * DIMD);
  unsigned short* wbIso  = allocH((size_t)DIMD * DIMD);

  // activations
  int*            idxBuf  = allocI((size_t)BT * NUM_TABLES);
  float*          x0      = allocF((size_t)BT * DIMD);
  unsigned short* x0bf    = allocH((size_t)BT * DIMD);
  float*          zg      = allocF((size_t)BT * SCAND);
  float*          zdt     = allocF((size_t)BT * SCAND);
  float*          zin     = allocF((size_t)BT * SCAND);
  float*          gatesB  = allocF((size_t)BT * SCAND);
  float*          driveB  = allocF((size_t)BT * SCAND);
  unsigned short* statesBf= allocH((size_t)BT * SCAND);
  float*          yBuf    = allocF((size_t)BT * DIMD);
  float*          x1      = allocF((size_t)BT * DIMD);
  unsigned short* x1bf    = allocH((size_t)BT * DIMD);
  float*          qkvBuf  = allocF((size_t)BT * 3 * DIMD);
  unsigned short* qbf     = allocH(NBHTD);
  unsigned short* kbf     = allocH(NBHTD);
  unsigned short* vbf     = allocH(NBHTD);
  unsigned short* attnObf = allocH((size_t)BT * DIMD);
  float*          attnP   = allocF((size_t)BT * DIMD);
  float*          x2      = allocF((size_t)BT * DIMD);
  unsigned short* x2bf    = allocH((size_t)BT * DIMD);
  unsigned short* Ibf     = allocH((size_t)BSZ * 64 * DIMD);
  float*          qdOut   = allocF((size_t)BSZ * 64 * DIMD);
  unsigned short* qdbf    = allocH(NBHMD);
  float*          kdOut   = allocF((size_t)BT * DIMD);
  unsigned short* kdbf    = allocH(NBHTD);
  float*          vdOut   = allocF((size_t)BT * DIMD);
  unsigned short* vdbf    = allocH(NBHTD);
  unsigned short* Hbf     = allocH((size_t)BSZ * 64 * DIMD);
  float*          kuOut   = allocF((size_t)BSZ * 64 * DIMD);
  unsigned short* kubf    = allocH(NBHMD);
  float*          vuOut   = allocF((size_t)BSZ * 64 * DIMD);
  unsigned short* vubf    = allocH(NBHMD);
  float*          quOut   = allocF((size_t)BT * DIMD);
  unsigned short* qubf    = allocH(NBHTD);
  unsigned short* O2bf    = allocH((size_t)BT * DIMD);
  float*          isoP    = allocF((size_t)BT * DIMD);

  auto wconv = [&](const float* w, unsigned short* dst, int n) {
    f32_to_bf16_kernel<<<(n + 255) / 256, 256, 0, stream>>>(w, dst, n);
  };
  auto gemm = [&](const unsigned short* A, const unsigned short* W, const float* bias,
                  float* C, int M, int N, int K) {
    gemm_bf16_wmma_kernel<<<dim3(N / 64, M / 128), dim3(256), 0, stream>>>(A, W, bias, C, M, N, K);
  };
  auto split = [&](const float* src, unsigned short* dst, int T, int srow, int coff) {
    int total = BSZ * HEADSN * T * HDN;
    split_bhtd_kernel<<<(total + 255) / 256, 256, 0, stream>>>(src, dst, T, srow, coff);
  };
  auto flash = [&](const unsigned short* Q, const unsigned short* Kp, const unsigned short* V,
                   unsigned short* O, int Tq, int Tk, int causal) {
    flash_attn_kernel<<<dim3(Tq / 16, HEADSN, BSZ), dim3(32), 0, stream>>>(
        Q, Kp, V, O, Tq, Tk, causal, 0.125f);
  };

  // --- weights to bf16 ---
  wconv(gate_W, wbGate, DIMD * SCAND);
  wconv(dt_W,   wbDt,   DIMD * SCAND);
  wconv(in_W,   wbIn,   DIMD * SCAND);
  wconv(scan_out_W, wbScan, SCAND * DIMD);
  wconv(qkv_W,  wbQkv,  DIMD * 3 * DIMD);
  wconv(attn_out_W, wbAttn, DIMD * DIMD);
  wconv(qd_W, wbQd, DIMD * DIMD);
  wconv(kd_W, wbKd, DIMD * DIMD);
  wconv(vd_W, wbVd, DIMD * DIMD);
  wconv(qu_W, wbQu, DIMD * DIMD);
  wconv(ku_W, wbKu, DIMD * DIMD);
  wconv(vu_W, wbVu, DIMD * DIMD);
  wconv(isab_out_W, wbIso, DIMD * DIMD);

  // --- hash embedding ---
  hash_idx_kernel<<<(BT + 255) / 256, 256, 0, stream>>>(tokens, idxBuf, BSZ, TLEN);
  gather_embed_kernel<<<(BT * DIMD + 255) / 256, 256, 0, stream>>>(
      tables, idxBuf, x0, x0bf, BT * DIMD);

  // --- gated scan ---
  gemm(x0bf, wbGate, gate_b, zg,  BT, SCAND, DIMD);
  gemm(x0bf, wbDt,   dt_b,   zdt, BT, SCAND, DIMD);
  gemm(x0bf, wbIn,   in_b,   zin, BT, SCAND, DIMD);
  scan_gates_kernel<<<(BT * SCAND + 255) / 256, 256, 0, stream>>>(
      zg, zdt, zin, gatesB, driveB, BT * SCAND);
  scan_seq_kernel<<<(BSZ * SCAND + 255) / 256, 256, 0, stream>>>(
      gatesB, driveB, statesBf, BSZ, TLEN, SCAND);
  gemm(statesBf, wbScan, scan_out_b, yBuf, BT, DIMD, SCAND);
  ln_fused_kernel<<<BT, 256, 0, stream>>>(x0, yBuf, scan_ln_g, scan_ln_b, x1, x1bf);

  // --- causal attention ---
  gemm(x1bf, wbQkv, qkv_b, qkvBuf, BT, 3 * DIMD, DIMD);
  split(qkvBuf, qbf, TLEN, 3 * DIMD, 0);
  split(qkvBuf, kbf, TLEN, 3 * DIMD, DIMD);
  split(qkvBuf, vbf, TLEN, 3 * DIMD, 2 * DIMD);
  flash(qbf, kbf, vbf, attnObf, TLEN, TLEN, 1);
  gemm(attnObf, wbAttn, attn_out_b, attnP, BT, DIMD, DIMD);
  ln_fused_kernel<<<BT, 256, 0, stream>>>(x1, attnP, attn_ln_g, attn_ln_b, x2, x2bf);

  // --- ISAB ---
  replicate_ind_kernel<<<(BSZ * 64 * DIMD + 255) / 256, 256, 0, stream>>>(
      inducing, Ibf, BSZ * 64 * DIMD);
  gemm(Ibf, wbQd, qd_b, qdOut, BSZ * 64, DIMD, DIMD);
  split(qdOut, qdbf, 64, DIMD, 0);
  gemm(x2bf, wbKd, kd_b, kdOut, BT, DIMD, DIMD);
  split(kdOut, kdbf, TLEN, DIMD, 0);
  gemm(x2bf, wbVd, vd_b, vdOut, BT, DIMD, DIMD);
  split(vdOut, vdbf, TLEN, DIMD, 0);
  flash(qdbf, kdbf, vdbf, Hbf, 64, TLEN, 0);          // H = mha(I, x, x)

  gemm(Hbf, wbKu, ku_b, kuOut, BSZ * 64, DIMD, DIMD);
  split(kuOut, kubf, 64, DIMD, 0);
  gemm(Hbf, wbVu, vu_b, vuOut, BSZ * 64, DIMD, DIMD);
  split(vuOut, vubf, 64, DIMD, 0);
  gemm(x2bf, wbQu, qu_b, quOut, BT, DIMD, DIMD);
  split(quOut, qubf, TLEN, DIMD, 0);
  flash(qubf, kubf, vubf, O2bf, TLEN, 64, 0);         // mha(x, H, H)

  gemm(O2bf, wbIso, isab_out_b, isoP, BT, DIMD, DIMD);
  ln_fused_kernel<<<BT, 256, 0, stream>>>(x2, isoP, isab_ln_g, isab_ln_b, out, (unsigned short*)nullptr);
}
